// PredictionHeadEdge_30356828848150
// MI455X (gfx1250) — matrix operand order repaired and checked by reading, not compiled, not run on hardware
//
#include <hip/hip_runtime.h>
#include <hip/hip_bf16.h>
#include <stdint.h>

#define N_  512
#define SD_ 256
#define VD_ 128
#define ED_ 128
#define E_  262144
#define NG_ 16
#define NA_ 16
#define NB_ 5

typedef __attribute__((ext_vector_type(16))) __bf16        v16bf;
typedef __attribute__((ext_vector_type(8)))  float         v8f;
typedef __attribute__((ext_vector_type(4)))  float         v4f;
typedef __attribute__((ext_vector_type(4)))  unsigned int  v4u;
typedef __attribute__((ext_vector_type(2)))  unsigned int  v2u;

union Frag { v4u q[2]; v16bf v; };

__device__ __forceinline__ unsigned short f2bf(float f) {
    union { float f; unsigned u; } x; x.f = f;
    unsigned r = x.u + 0x7FFFu + ((x.u >> 16) & 1u);   // round-to-nearest-even
    return (unsigned short)(r >> 16);
}
__device__ __forceinline__ float bf2f(unsigned short h) {
    union { unsigned u; float f; } x; x.u = ((unsigned)h) << 16;
    return x.f;
}
__device__ __forceinline__ float bflo(unsigned u) {
    union { unsigned u; float f; } x; x.u = u << 16; return x.f;
}
__device__ __forceinline__ float bfhi(unsigned u) {
    union { unsigned u; float f; } x; x.u = u & 0xffff0000u; return x.f;
}
__device__ __forceinline__ unsigned packbf2(float lo, float hi) {
    return (unsigned)f2bf(lo) | ((unsigned)f2bf(hi) << 16);
}
// add two packed-bf16 pairs in f32, repack
__device__ __forceinline__ unsigned addbf2(unsigned a, unsigned b) {
    return packbf2(bflo(a) + bflo(b), bfhi(a) + bfhi(b));
}
__device__ __forceinline__ float silu_f(float x) { return x / (1.0f + __expf(-x)); }

// ---------------------------------------------------------------- init / scatter
__global__ void k_init(int* idxmap) {
    int t = blockIdx.x * blockDim.x + threadIdx.x;
    if (t < N_ * N_) idxmap[t] = -1;
}

// sequential .set semantics == max edge id wins for duplicate (j,i)
__global__ void k_scatter(const int* __restrict__ eidx, int* __restrict__ idxmap) {
    int k = blockIdx.x * blockDim.x + threadIdx.x;
    if (k < E_) {
        int j = eidx[k];
        int i = eidx[E_ + k];
        atomicMax(&idxmap[j * N_ + i], k);
    }
}

// ---------------------------------------------------------------- node stage 1: s2 = silu(s@W_sm+b), raw coords
__global__ void k_node1(const float* __restrict__ s, const float* __restrict__ v,
                        const float* __restrict__ p, const float* __restrict__ W_sm,
                        const float* __restrict__ b_sm, const float* __restrict__ W_c,
                        float* __restrict__ s2, float* __restrict__ rawc) {
    int base = blockIdx.x * 16;
    int tid  = threadIdx.x;
    for (int t = tid; t < 16 * SD_; t += 256) {
        int row = base + (t >> 8);
        int n   = t & 255;
        float acc = b_sm[n];
        const float* srow = s + row * SD_;
        for (int k = 0; k < SD_; ++k) acc = fmaf(srow[k], W_sm[k * SD_ + n], acc);
        s2[row * SD_ + n] = silu_f(acc);
    }
    if (tid < 48) {
        int row = base + tid / 3, x = tid % 3;
        float acc = p[row * 3 + x];
        const float* vrow = v + (row * 3 + x) * VD_;
        for (int k = 0; k < VD_; ++k) acc = fmaf(vrow[k], W_c[k], acc);
        rawc[row * 3 + x] = acc;
    }
}

// ---------------------------------------------------------------- deterministic per-graph means (NG=16 tiny)
__global__ void k_center(const float* __restrict__ rawc, const int* __restrict__ batch,
                         float* __restrict__ means) {
    int t = threadIdx.x;
    if (t < NG_ * 3) {
        int g = t / 3, x = t % 3;
        float sum = 0.f, cnt = 0.f;
        for (int r = 0; r < N_; ++r)
            if (batch[r] == g) { sum += rawc[r * 3 + x]; cnt += 1.f; }
        means[t] = sum / fmaxf(cnt, 1.f);
    }
}

// ---------------------------------------------------------------- node stage 2: Spb = bf16(s2@W_b0[:SD]), atoms, coords
__global__ void k_node2(const float* __restrict__ s2, const float* __restrict__ rawc,
                        const float* __restrict__ means, const int* __restrict__ batch,
                        const float* __restrict__ W_b0, const float* __restrict__ W_a,
                        const float* __restrict__ b_a, unsigned short* __restrict__ Spb,
                        float* __restrict__ out_coords, float* __restrict__ out_atoms) {
    int base = blockIdx.x * 16;
    int tid  = threadIdx.x;
    for (int t = tid; t < 16 * SD_; t += 256) {
        int row = base + (t >> 8);
        int n   = t & 255;
        float acc = 0.f;
        const float* srow = s2 + row * SD_;
        for (int k = 0; k < SD_; ++k) acc = fmaf(srow[k], W_b0[k * SD_ + n], acc);
        Spb[row * SD_ + n] = f2bf(acc);
    }
    if (tid < 16 * NA_) {
        int row = base + (tid >> 4), na = tid & 15;
        float acc = b_a[na];
        const float* srow = s2 + row * SD_;
        for (int k = 0; k < SD_; ++k) acc = fmaf(srow[k], W_a[k * NA_ + na], acc);
        out_atoms[row * NA_ + na] = acc;
    }
    if (tid < 48) {
        int row = base + tid / 3, x = tid % 3;
        out_coords[row * 3 + x] = rawc[row * 3 + x] - means[batch[row] * 3 + x];
    }
}

// ---------------------------------------------------------------- weight folding:
// Ut[n][k] = bf16((W_bm@W_b0[:SD])[k][n]), c0 = b_b0 + b_bm@W_b0[:SD],
// Wb1t[n][k] = bf16(W_b1[k][n]) zero-padded to 16 output cols
__global__ void k_prep(const float* __restrict__ W_bm, const float* __restrict__ b_bm,
                       const float* __restrict__ W_b0, const float* __restrict__ b_b0,
                       const float* __restrict__ W_b1,
                       unsigned short* __restrict__ Ut, float* __restrict__ c0,
                       unsigned short* __restrict__ Wb1t) {
    int idx = blockIdx.x * 512 + threadIdx.x;
    for (int rep = 0; rep < 2; ++rep, idx += 256) {
        int n = idx >> 7, k = idx & 127;
        float acc = 0.f;
        for (int c = 0; c < SD_; ++c) acc = fmaf(W_bm[k * SD_ + c], W_b0[c * SD_ + n], acc);
        Ut[n * ED_ + k] = f2bf(acc);
    }
    if (blockIdx.x == 0) {
        int n = threadIdx.x;
        float acc = b_b0[n];
        for (int c = 0; c < SD_; ++c) acc = fmaf(b_bm[c], W_b0[c * SD_ + n], acc);
        c0[n] = acc;
    }
    if (blockIdx.x < 8) {
        int idx2 = blockIdx.x * 512 + threadIdx.x;
        for (int rep = 0; rep < 2; ++rep, idx2 += 256) {
            int n = idx2 >> 8, k = idx2 & 255;
            float w = (n < NB_) ? W_b1[k * NB_ + n] : 0.f;
            Wb1t[n * SD_ + k] = f2bf(w);
        }
    }
}

// ---------------------------------------------------------------- edge kernel: 128 edges/block, bf16 WMMA
__global__ __launch_bounds__(256) void k_edge(
        const float* __restrict__ e, const int* __restrict__ eidx,
        const int* __restrict__ idxmap, const float* __restrict__ coords,
        const unsigned short* __restrict__ Spb, const unsigned short* __restrict__ Ut,
        const float* __restrict__ c0, const float* __restrict__ W_b0,
        const unsigned short* __restrict__ Wb1t, const float* __restrict__ b_b1,
        float* __restrict__ out_bonds) {
    extern __shared__ char smem[];
    unsigned short* hbuf  = (unsigned short*)smem;              // [128][256] bf16: spair -> silu(h)
    unsigned short* esym  = (unsigned short*)(smem + 65536);    // [128][128] bf16
    float*          edist = (float*)(smem + 98304);             // [128]
    int*            ei    = (int*)(smem + 98816);
    int*            ej    = (int*)(smem + 99328);
    int*            ea    = (int*)(smem + 99840);
    int*            eb    = (int*)(smem + 100352);
    float*          bb1s  = (float*)(smem + 100864);            // [8]

    int tid = threadIdx.x;

    // ---- async Global->LDS: bias vector, drained before the output stage
    if (tid < NB_) {
        unsigned ldsoff = (unsigned)(unsigned long long)(const void*)(bb1s + tid);
        unsigned long long ga = (unsigned long long)(const void*)(b_b1 + tid);
        asm volatile("global_load_async_to_lds_b32 %0, %1, off"
                     :: "v"(ldsoff), "v"(ga) : "memory");
    }

    // ---- stage A: per-edge metadata (+ prefetch scattered e rows into cache)
    if (tid < 128) {
        int g = blockIdx.x * 128 + tid;
        int j = eidx[g];
        int i = eidx[E_ + g];
        ei[tid] = i; ej[tid] = j;
        int a = idxmap[j * N_ + i];
        int b = idxmap[i * N_ + j];
        ea[tid] = a; eb[tid] = b;
        if (a >= 0) __builtin_prefetch(e + (size_t)a * ED_, 0, 3);
        if (b >= 0) __builtin_prefetch(e + (size_t)b * ED_, 0, 3);
        float dx = coords[i * 3 + 0] - coords[j * 3 + 0];
        float dy = coords[i * 3 + 1] - coords[j * 3 + 1];
        float dz = coords[i * 3 + 2] - coords[j * 3 + 2];
        edist[tid] = dx * dx + dy * dy + dz * dz;
    }
    __syncthreads();

    // ---- stage A: gather e_sym = 0.5*(e[a]+e[b]) -> LDS bf16 (packed 8B stores)
    v4f z4 = {0.f, 0.f, 0.f, 0.f};
    for (int t = tid; t < 128 * 32; t += 256) {
        int m  = t >> 5;
        int c4 = (t & 31) << 2;
        int a = ea[m], b = eb[m];
        v4f va = (a >= 0) ? *(const v4f*)(e + (size_t)a * ED_ + c4) : z4;
        v4f vb = (b >= 0) ? *(const v4f*)(e + (size_t)b * ED_ + c4) : z4;
        v4f vs = 0.5f * (va + vb);
        v2u pk;
        pk.x = packbf2(vs.x, vs.y);
        pk.y = packbf2(vs.z, vs.w);
        *(v2u*)(esym + m * ED_ + c4) = pk;
    }
    // ---- stage A: spair = Spb[i]+Spb[j] (bf16 source, halved gather traffic) -> hbuf
    for (int t = tid; t < 128 * 32; t += 256) {
        int m  = t >> 5;
        int c8 = (t & 31) << 3;
        int i = ei[m], j = ej[m];
        v4u a = *(const v4u*)(Spb + (size_t)i * SD_ + c8);
        v4u b = *(const v4u*)(Spb + (size_t)j * SD_ + c8);
        v4u o;
        o.x = addbf2(a.x, b.x);
        o.y = addbf2(a.y, b.y);
        o.z = addbf2(a.z, b.z);
        o.w = addbf2(a.w, b.w);
        *(v4u*)(hbuf + m * SD_ + c8) = o;
    }
    __syncthreads();

    // ---- stage B: WMMA  acc = esym(bf16) @ U(bf16), f32 accumulate
    int wave = tid >> 5, lane = tid & 31;
    int lm = lane & 15, lh = lane >> 4;
    int n0 = wave * 32 + lm;                 // this wave owns output cols [wave*32, wave*32+32)
    v8f acc[8][2] = {};                      // 8 M-tiles x 2 N-tiles

#pragma unroll
    for (int kb = 0; kb < ED_; kb += 32) {
        Frag fb0, fb1;
        // B operand in this code's contiguous k order: lane = N, half-waves split K 0-15 / 16-31
        const v4u* pb0 = (const v4u*)(Ut + (size_t)n0 * ED_ + kb + lh * 16);
        fb0.q[0] = pb0[0]; fb0.q[1] = pb0[1];
        const v4u* pb1 = (const v4u*)(Ut + (size_t)(n0 + 16) * ED_ + kb + lh * 16);
        fb1.q[0] = pb1[0]; fb1.q[1] = pb1[1];
#pragma unroll
        for (int mt = 0; mt < 8; ++mt) {
            Frag fa;
            // A operand in this code's contiguous k order: lane = M; half-waves split K 0-15 / 16-31 (matches B)
            const v4u* pa0 = (const v4u*)(esym + (mt * 16 + lm) * ED_ + kb + lh * 16);
            const v4u* pa1 = (const v4u*)(esym + (mt * 16 + lm) * ED_ + kb + lh * 16 + 8);
            fa.q[0] = pa0[0];
            fa.q[1] = pa1[0];
            acc[mt][0] = __builtin_amdgcn_wmma_f32_16x16x32_bf16(
                false, fa.v, false, fb0.v, (short)0, acc[mt][0], false, false);
            acc[mt][1] = __builtin_amdgcn_wmma_f32_16x16x32_bf16(
                false, fa.v, false, fb1.v, (short)0, acc[mt][1], false, false);
        }
    }

    // ---- stage C: h = acc + spair + d*w_d + c0 ; silu ; write bf16 in place
    float wd0 = W_b0[SD_ * SD_ + n0];        // W_b0 row SD (distance column weight)
    float wd1 = W_b0[SD_ * SD_ + n0 + 16];
    float c00 = c0[n0], c01 = c0[n0 + 16];
#pragma unroll
    for (int mt = 0; mt < 8; ++mt) {
#pragma unroll
        for (int r = 0; r < 8; ++r) {
            int m = mt * 16 + r + lh * 8;    // C/D layout: lanes 0-15 -> M=r, 16-31 -> M=r+8
            float dm = edist[m];
            unsigned short* h0p = hbuf + m * SD_ + n0;
            float h0 = acc[mt][0][r] + bf2f(*h0p) + dm * wd0 + c00;
            *h0p = f2bf(silu_f(h0));
            unsigned short* h1p = h0p + 16;
            float h1 = acc[mt][1][r] + bf2f(*h1p) + dm * wd1 + c01;
            *h1p = f2bf(silu_f(h1));
        }
    }
    // drain async bias load, then make all LDS writes visible
    asm volatile("s_wait_asynccnt 0x0" ::: "memory");
    __syncthreads();

    // ---- stage D: bonds = silu(h) @ W_b1 + b_b1 via WMMA (1 M-tile per wave, K=256)
    v8f accd = {};
#pragma unroll
    for (int kb = 0; kb < SD_; kb += 32) {
        Frag fb;
        const v4u* pb = (const v4u*)(Wb1t + (size_t)lm * SD_ + kb + lh * 16);
        fb.q[0] = pb[0]; fb.q[1] = pb[1];
        Frag fa;
        const v4u* pa0 = (const v4u*)(hbuf + (wave * 16 + lm) * SD_ + kb + lh * 16);
        const v4u* pa1 = (const v4u*)(hbuf + (wave * 16 + lm) * SD_ + kb + lh * 16 + 8);
        fa.q[0] = pa0[0];
        fa.q[1] = pa1[0];
        accd = __builtin_amdgcn_wmma_f32_16x16x32_bf16(
            false, fa.v, false, fb.v, (short)0, accd, false, false);
    }
    if (lm < NB_) {
        float bias = bb1s[lm];
#pragma unroll
        for (int r = 0; r < 8; ++r) {
            int m = wave * 16 + r + lh * 8;
            out_bonds[(size_t)(blockIdx.x * 128 + m) * NB_ + lm] = accd[r] + bias;
        }
    }
}

// ---------------------------------------------------------------- launch
extern "C" void kernel_launch(void* const* d_in, const int* in_sizes, int n_in,
                              void* d_out, int out_size, void* d_ws, size_t ws_size,
                              hipStream_t stream) {
    (void)in_sizes; (void)n_in; (void)out_size; (void)ws_size;

    const float* s    = (const float*)d_in[0];
    const float* v    = (const float*)d_in[1];
    const float* p    = (const float*)d_in[2];
    const float* e    = (const float*)d_in[3];
    const int*   batch= (const int*)d_in[4];
    const int*   eidx = (const int*)d_in[5];
    const float* W_sm = (const float*)d_in[6];
    const float* b_sm = (const float*)d_in[7];
    const float* W_bm = (const float*)d_in[8];
    const float* b_bm = (const float*)d_in[9];
    const float* W_b0 = (const float*)d_in[10];
    const float* b_b0 = (const float*)d_in[11];
    const float* W_b1 = (const float*)d_in[12];
    const float* b_b1 = (const float*)d_in[13];
    const float* W_c  = (const float*)d_in[14];
    const float* W_a  = (const float*)d_in[15];
    const float* b_a  = (const float*)d_in[16];

    float* out_coords = (float*)d_out;                  // [512,3]
    float* out_atoms  = out_coords + N_ * 3;            // [512,16]
    float* out_bonds  = out_atoms + N_ * NA_;           // [E,5]

    // workspace layout (~1.83 MB)
    char* ws = (char*)d_ws;
    int*            idxmap = (int*)ws;                          // 1,048,576 B
    float*          s2     = (float*)(ws + 1048576);            //   524,288 B
    unsigned short* Spb    = (unsigned short*)(ws + 1572864);   //   262,144 B
    float*          rawc   = (float*)(ws + 1835008);            //     6,144 B
    float*          means  = (float*)(ws + 1841152);            //       192 B
    float*          c0     = (float*)(ws + 1841408);            //     1,024 B
    unsigned short* Ut     = (unsigned short*)(ws + 1842432);   //    65,536 B
    unsigned short* Wb1t   = (unsigned short*)(ws + 1907968);   //     8,192 B

    k_init   <<<(N_ * N_) / 256, 256, 0, stream>>>(idxmap);
    k_scatter<<<E_ / 256,        256, 0, stream>>>(eidx, idxmap);
    k_node1  <<<32,              256, 0, stream>>>(s, v, p, W_sm, b_sm, W_c, s2, rawc);
    k_center <<<1,                64, 0, stream>>>(rawc, batch, means);
    k_node2  <<<32,              256, 0, stream>>>(s2, rawc, means, batch, W_b0, W_a, b_a,
                                                   Spb, out_coords, out_atoms);
    k_prep   <<<64,              256, 0, stream>>>(W_bm, b_bm, W_b0, b_b0, W_b1, Ut, c0, Wb1t);

    size_t smem = 100896;  // hbuf 64K + esym 32K + per-edge scalars + bias
    k_edge<<<E_ / 128, 256, smem, stream>>>(e, eidx, idxmap, out_coords, Spb, Ut, c0,
                                            W_b0, Wb1t, b_b1, out_bonds);
}
